// MyLlamaAttention_31593779429914
// MI455X (gfx1250) — matrix operand order repaired and checked by reading, not compiled, not run on hardware
//
#include <hip/hip_runtime.h>
#include <math.h>

typedef __attribute__((ext_vector_type(16))) _Float16 v16h;
typedef __attribute__((ext_vector_type(8)))  float    v8f;
typedef int async_v4i __attribute__((vector_size(16)));   // b128 unit for async DMA

#define B_    2
#define S_    2048
#define H_    4096
#define HQ    32
#define KVH   8
#define D_    128
#define MROWS (B_ * S_)

static __device__ __forceinline__ v8f wmma_f16(v16h a, v16h b, v8f c) {
  // lowers to v_wmma_f32_16x16x32_f16
  return __builtin_amdgcn_wmma_f32_16x16x32_f16(false, a, false, b, (short)0, c,
                                                false, false);
}

// ---------------------------------------------------------------------------
// GEMM: C[M,N] = A[M,K] * W[N,K]^T  (f16 WMMA math, f32 accumulate)
// TIN:  element type of A (float or _Float16); W is always float.
// TOUT: element type of C (float or _Float16).
// Block: 256 threads (8 waves). Tile: 64(M) x 128(N), K-step 32.
// Wave grid 2(M) x 4(N); each wave owns 32x32 C = 2 A-frags x 2 B-frags
// -> 4 WMMAs per 8 ds_load_b128. Double-buffered LDS + reg prefetch +
// global_prefetch two tiles ahead.
// ---------------------------------------------------------------------------
template <typename TIN, typename TOUT>
__global__ __launch_bounds__(256)
void gemm_xwt(const TIN* __restrict__ A, const float* __restrict__ W,
              TOUT* __restrict__ C, int M, int N, int K)
{
  __shared__ _Float16 As[2][64][40];    // [buf][m][k] (padded)
  __shared__ _Float16 Bs[2][128][40];   // [buf][n][k] (padded)

  const int m0   = blockIdx.y * 64;
  const int n0   = blockIdx.x * 128;
  const int tid  = threadIdx.x;
  const int lane = tid & 31;
  const int wave = tid >> 5;            // 0..7
  const int wm   = (wave & 1) * 32;     // wave row offset (2 waves over M)
  const int wn   = (wave >> 1) * 32;    // wave col offset (4 waves over N)
  const int lrow = lane & 15;
  const int lhi  = lane >> 4;

  // Staging: A 64x32 (8 elems/thread), B 128x32 (16 elems/thread)
  const int aR = tid >> 2;              // 0..63
  const int aC = (tid & 3) * 8;         // 0,8,16,24
  const int bR = tid >> 1;              // 0..127
  const int bC = (tid & 1) * 16;        // 0,16
  const TIN*   aPtr = A + (size_t)(m0 + aR) * K + aC;
  const float* bPtr = W + (size_t)(n0 + bR) * K + bC;

  v8f z = {};
  v8f acc[2][2];
  acc[0][0] = z; acc[0][1] = z; acc[1][0] = z; acc[1][1] = z;

  const int NT = K / 32;

  // Stage tile 0 directly into LDS buffer 0
  #pragma unroll
  for (int j = 0; j < 8; ++j)  As[0][aR][aC + j] = (_Float16)aPtr[j];
  #pragma unroll
  for (int j = 0; j < 16; ++j) Bs[0][bR][bC + j] = (_Float16)bPtr[j];
  __syncthreads();

  for (int kt = 0; kt < NT; ++kt) {
    const int buf = kt & 1;

    // Register-prefetch next global tile (overlaps with WMMA below)
    TIN   aReg[8];
    float bReg[16];
    if (kt + 1 < NT) {
      const TIN*   ap = aPtr + (kt + 1) * 32;
      const float* bp = bPtr + (kt + 1) * 32;
      #pragma unroll
      for (int j = 0; j < 8; ++j)  aReg[j] = ap[j];
      #pragma unroll
      for (int j = 0; j < 16; ++j) bReg[j] = bp[j];
    }
    if (kt + 2 < NT) {                 // L2 prefetch two tiles ahead
      __builtin_prefetch(aPtr + (kt + 2) * 32, 0, 1);
      __builtin_prefetch(bPtr + (kt + 2) * 32, 0, 1);
    }

    // Fragments + 4 WMMAs
    v16h af[2], bf[2];
    #pragma unroll
    for (int mi = 0; mi < 2; ++mi)
      #pragma unroll
      for (int e = 0; e < 8; ++e) {
        af[mi][e]     = As[buf][wm + mi * 16 + lrow][lhi * 16 + e];
        af[mi][e + 8] = As[buf][wm + mi * 16 + lrow][lhi * 16 + 8 + e];
      }
    #pragma unroll
    for (int ni = 0; ni < 2; ++ni)
      #pragma unroll
      for (int e = 0; e < 16; ++e)
        bf[ni][e] = Bs[buf][wn + ni * 16 + lrow][lhi * 16 + e];

    acc[0][0] = wmma_f16(af[0], bf[0], acc[0][0]);
    acc[0][1] = wmma_f16(af[0], bf[1], acc[0][1]);
    acc[1][0] = wmma_f16(af[1], bf[0], acc[1][0]);
    acc[1][1] = wmma_f16(af[1], bf[1], acc[1][1]);

    // Store prefetched tile into the other buffer
    if (kt + 1 < NT) {
      const int nb = (kt + 1) & 1;
      #pragma unroll
      for (int j = 0; j < 8; ++j)  As[nb][aR][aC + j] = (_Float16)aReg[j];
      #pragma unroll
      for (int j = 0; j < 16; ++j) Bs[nb][bR][bC + j] = (_Float16)bReg[j];
    }
    __syncthreads();
  }

  // C/D layout: col = lrow, rows = lhi*8 + r
  #pragma unroll
  for (int mi = 0; mi < 2; ++mi)
    #pragma unroll
    for (int ni = 0; ni < 2; ++ni)
      #pragma unroll
      for (int r = 0; r < 8; ++r) {
        int row = m0 + wm + mi * 16 + lhi * 8 + r;
        int col = n0 + wn + ni * 16 + lrow;
        C[(size_t)row * N + col] = (TOUT)acc[mi][ni][r];
      }
}

// ---------------------------------------------------------------------------
// RoPE applied in place to q (B,S,HQ,D) and k (B,S,KVH,D), f16 storage.
// ---------------------------------------------------------------------------
__global__ void rope_kernel(_Float16* __restrict__ q, _Float16* __restrict__ k,
                            const int* __restrict__ pos)
{
  int idx = blockIdx.x * blockDim.x + threadIdx.x;
  const int total = B_ * S_ * (HQ + KVH) * (D_ / 2);
  if (idx >= total) return;
  int i = idx % (D_ / 2); idx /= (D_ / 2);
  int h = idx % (HQ + KVH); idx /= (HQ + KVH);
  int s = idx % S_;
  int b = idx / S_;

  float p   = (float)pos[b * S_ + s];
  float inv = __powf(10000.0f, -2.0f * (float)i / (float)D_);
  float ang = p * inv;
  float cs  = __cosf(ang), sn = __sinf(ang);

  _Float16* base;
  if (h < HQ) base = q + ((size_t)(b * S_ + s) * HQ + h) * D_;
  else        base = k + ((size_t)(b * S_ + s) * KVH + (h - HQ)) * D_;

  float x1 = (float)base[i], x2 = (float)base[i + 64];
  base[i]      = (_Float16)(x1 * cs - x2 * sn);
  base[i + 64] = (_Float16)(x2 * cs + x1 * sn);
}

// ---------------------------------------------------------------------------
// Flash attention: 4 waves/block, 64 query rows, shared K/V LDS tiles.
// q,k,v are f16 (B,S,heads,D). K tile staged [key][d] via async-to-LDS DMA
// (ASYNCcnt) when the builtin exists; V staged transposed [d][key] so PV
// B-fragments are contiguous b128 LDS reads. 32 keys per step.
// ---------------------------------------------------------------------------
__global__ __launch_bounds__(128)
void flash_attn(const _Float16* __restrict__ q, const _Float16* __restrict__ k,
                const _Float16* __restrict__ v, _Float16* __restrict__ o)
{
  __shared__ _Float16 Kt[32][136];     // [key][d]  row stride 272B (16B aligned)
  __shared__ _Float16 Vt[128][40];     // [d][key]  (padded)
  __shared__ _Float16 P[4][16][40];    // per-wave probability staging

  int bid = blockIdx.x;
  int qb  = bid % (S_ / 64); bid /= (S_ / 64);
  int h   = bid % HQ;
  int b   = bid / HQ;
  int kvh = h >> 2;                    // n_rep = 4

  const int tid  = threadIdx.x;
  const int wave = tid >> 5;
  const int lane = tid & 31;
  const int lrow = lane & 15;
  const int lhi  = lane >> 4;
  const int qt   = qb * 64 + wave * 16;            // this wave's first query
  const float scale = 0.08838834764831845f;        // 1/sqrt(128)

  // Q as four 16x32 f16 A-fragments (pre-scaled)
  v16h qa[4];
  const _Float16* qrow = q + ((size_t)(b * S_ + qt + lrow) * HQ + h) * D_;
  #pragma unroll
  for (int kc = 0; kc < 4; ++kc)
    #pragma unroll
    for (int e = 0; e < 8; ++e) {
      qa[kc][e]     = (_Float16)((float)qrow[kc * 32 + lhi * 16 + e] * scale);
      qa[kc][e + 8] = (_Float16)((float)qrow[kc * 32 + lhi * 16 + 8 + e] * scale);
    }

  v8f z = {};
  v8f oacc[8];
  #pragma unroll
  for (int c = 0; c < 8; ++c) oacc[c] = z;
  float mrow[8], lsum[8];
  #pragma unroll
  for (int r = 0; r < 8; ++r) { mrow[r] = -3.0e38f; lsum[r] = 0.0f; }

  const int kendw = qt + 16;           // wave causal horizon (exclusive)
  const int kendb = qb * 64 + 64;      // block horizon

  for (int j0 = 0; j0 < kendb; j0 += 32) {
    // ---- cooperative staging: 32 keys x 128 d of f16.
    // b128 unit u: key = u>>4, d = (u&15)*8 ; 512 units over 128 threads.
    bool used_async = false;
    #pragma unroll
    for (int i = 0; i < 4; ++i) {
      int u   = i * 128 + tid;
      int key = u >> 4;
      int dc  = (u & 15) * 8;
      const _Float16* kp = k + ((size_t)(b * S_ + j0 + key) * KVH + kvh) * D_ + dc;
      const _Float16* vp = v + ((size_t)(b * S_ + j0 + key) * KVH + kvh) * D_ + dc;
#if __has_builtin(__builtin_amdgcn_global_load_async_to_lds_b128)
      // K tile: layout-preserving 16B DMA, tracked by ASYNCcnt
      __builtin_amdgcn_global_load_async_to_lds_b128(
          (__attribute__((address_space(1))) async_v4i*)kp,
          (__attribute__((address_space(3))) async_v4i*)(&Kt[key][dc]),
          0, 0);
      used_async = true;
#else
      #pragma unroll
      for (int j = 0; j < 8; ++j) Kt[key][dc + j] = kp[j];
#endif
      // V tile: transposed store (DMA cannot transpose)
      #pragma unroll
      for (int j = 0; j < 8; ++j) Vt[dc + j][key] = vp[j];
    }
    if (used_async) {
#if __has_builtin(__builtin_amdgcn_s_wait_asynccnt)
      __builtin_amdgcn_s_wait_asynccnt(0);
#else
      asm volatile("s_wait_asynccnt 0x0" ::: "memory");
#endif
    }
    __syncthreads();

    if (j0 < kendw) {
      float pv[2][8];
      float rmax[8];
      #pragma unroll
      for (int r = 0; r < 8; ++r) rmax[r] = -3.0e38f;

      // ---- S = Q * K^T for two 16-key tiles (8 WMMAs)
      #pragma unroll
      for (int kk = 0; kk < 2; ++kk) {
        v8f sfrag = z;
        int keyl = kk * 16 + lrow;       // B-frag col = key in tile
        #pragma unroll
        for (int kc = 0; kc < 4; ++kc) {
          v16h kb;
          #pragma unroll
          for (int e = 0; e < 16; ++e)
            kb[e] = Kt[keyl][kc * 32 + lhi * 16 + e];
          sfrag = wmma_f16(qa[kc], kb, sfrag);
        }
        int key = j0 + keyl;
        #pragma unroll
        for (int r = 0; r < 8; ++r) {
          int qidx = qt + lhi * 8 + r;
          float sv = (key <= qidx) ? sfrag[r] : -1.0e9f;
          pv[kk][r] = sv;
          rmax[r] = fmaxf(rmax[r], sv);
        }
      }

      // ---- online softmax (row reductions across 16-lane half-groups)
      #pragma unroll
      for (int r = 0; r < 8; ++r) {
        float rm = rmax[r];
        rm = fmaxf(rm, __shfl_xor(rm, 1, 32));
        rm = fmaxf(rm, __shfl_xor(rm, 2, 32));
        rm = fmaxf(rm, __shfl_xor(rm, 4, 32));
        rm = fmaxf(rm, __shfl_xor(rm, 8, 32));
        float mnew  = fmaxf(mrow[r], rm);
        float alpha = __expf(mrow[r] - mnew);
        mrow[r] = mnew;
        #pragma unroll
        for (int c = 0; c < 8; ++c) oacc[c][r] *= alpha;
        float ps = 0.0f;
        #pragma unroll
        for (int kk = 0; kk < 2; ++kk) {
          float e = __expf(pv[kk][r] - mnew);
          pv[kk][r] = e;
          ps += e;
        }
        ps += __shfl_xor(ps, 1, 32);
        ps += __shfl_xor(ps, 2, 32);
        ps += __shfl_xor(ps, 4, 32);
        ps += __shfl_xor(ps, 8, 32);
        lsum[r] = lsum[r] * alpha + ps;
      }

      // ---- transpose P (C-layout -> A-layout) via per-wave LDS slab.
      // Same-wave LDS RAW: in-order DS pipe, only need dscnt drain.
      #pragma unroll
      for (int kk = 0; kk < 2; ++kk)
        #pragma unroll
        for (int r = 0; r < 8; ++r)
          P[wave][lhi * 8 + r][kk * 16 + lrow] = (_Float16)pv[kk][r];
      asm volatile("s_wait_dscnt 0x0" ::: "memory");

      v16h pa;
      #pragma unroll
      for (int e = 0; e < 8; ++e) {
        pa[e]     = P[wave][lrow][lhi * 16 + e];
        pa[e + 8] = P[wave][lrow][lhi * 16 + 8 + e];
      }

      // ---- O += P(16x32) * V(32x128), eight d-chunks (8 WMMAs)
      #pragma unroll
      for (int c = 0; c < 8; ++c) {
        v16h vb;
        #pragma unroll
        for (int e = 0; e < 16; ++e)
          vb[e] = Vt[c * 16 + lrow][lhi * 16 + e];
        oacc[c] = wmma_f16(pa, vb, oacc[c]);
      }
    }
    __syncthreads();   // protect Kt/Vt before next staging round
  }

  // ---- normalize and write (B,S,HQ,D) f16
  #pragma unroll
  for (int c = 0; c < 8; ++c)
    #pragma unroll
    for (int r = 0; r < 8; ++r) {
      int row = qt + lhi * 8 + r;
      float val = oacc[c][r] / lsum[r];
      o[((size_t)(b * S_ + row) * HQ + h) * D_ + c * 16 + lrow] = (_Float16)val;
    }
}

// ---------------------------------------------------------------------------
extern "C" void kernel_launch(void* const* d_in, const int* in_sizes, int n_in,
                              void* d_out, int out_size, void* d_ws, size_t ws_size,
                              hipStream_t stream) {
  const float* hs  = (const float*)d_in[0];
  const int*   pos = (const int*)  d_in[1];
  const float* Wq  = (const float*)d_in[2];
  const float* Wk  = (const float*)d_in[3];
  const float* Wv  = (const float*)d_in[4];
  const float* Wo  = (const float*)d_in[5];
  float* out = (float*)d_out;

  // f16 intermediates: q(32MB) k(8MB) v(8MB) attn(32MB)
  _Float16* q    = (_Float16*)d_ws;                   // MROWS * 4096
  _Float16* kbuf = q    + (size_t)MROWS * H_;         // MROWS * 1024
  _Float16* vbuf = kbuf + (size_t)MROWS * KVH * D_;   // MROWS * 1024
  _Float16* attn = vbuf + (size_t)MROWS * KVH * D_;   // MROWS * 4096

  dim3 blk(256);
  gemm_xwt<float, _Float16>
      <<<dim3(H_ / 128, MROWS / 64), blk, 0, stream>>>(hs, Wq, q, MROWS, H_, H_);
  gemm_xwt<float, _Float16>
      <<<dim3((KVH * D_) / 128, MROWS / 64), blk, 0, stream>>>(hs, Wk, kbuf, MROWS, KVH * D_, H_);
  gemm_xwt<float, _Float16>
      <<<dim3((KVH * D_) / 128, MROWS / 64), blk, 0, stream>>>(hs, Wv, vbuf, MROWS, KVH * D_, H_);

  int total = B_ * S_ * (HQ + KVH) * (D_ / 2);
  rope_kernel<<<(total + 255) / 256, 256, 0, stream>>>(q, kbuf, pos);

  flash_attn<<<B_ * HQ * (S_ / 64), 128, 0, stream>>>(q, kbuf, vbuf, attn);

  gemm_xwt<_Float16, float>
      <<<dim3(H_ / 128, MROWS / 64), blk, 0, stream>>>(attn, Wo, out, MROWS, H_, H_);
}
